// SSM_Manipulation_fusion_24876450578938
// MI455X (gfx1250) — matrix operand order repaired and checked by reading, not compiled, not run on hardware
//
#include <hip/hip_runtime.h>
#include <hip/hip_bf16.h>

// ---------------------------------------------------------------------------
// SSM (Mamba-style) fused pipeline for gfx1250 (MI455X), wave32 + WMMA bf16.
//   B=4, D=1024, L=4096, N=16, R=64, W=4
// Pipeline:
//   1) causal dwconv+SiLU (x,mask,info) -> u,m,inf stored transposed (B,L,D)
//   2) d_low = delta_w @ m          (WMMA bf16, out bf16 (B,L,R))
//   3) dt    = softplus(dt_w @ d_low + dt_b)   (WMMA, out f32 (B,L,D))
//   4) Bm    = B_w @ inf ; Cm = C_w @ u        (WMMA, out f32 (B,L,N))
//   5) chunked selective scan (3 passes, f32)  -> y bf16 (B,L,D)
//   6) out   = adj_w @ y + adj_b               (WMMA, out f32 (B,D,L))
// ---------------------------------------------------------------------------

typedef __bf16 bf16;
typedef __attribute__((ext_vector_type(16))) __bf16 v16bf;
typedef __attribute__((ext_vector_type(8)))  __bf16 v8bf;
typedef __attribute__((ext_vector_type(8)))  float  v8f;

#define BB 4
#define DD 1024
#define LL 4096
#define NN 16
#define RR 64
#define WW 4
#define LC 64              // scan chunk length
#define NCH (LL / LC)      // 64 chunks

// ---------------------------------------------------------------------------
// f32 -> bf16 conversion (weights)
// ---------------------------------------------------------------------------
__global__ __launch_bounds__(256)
void f32_to_bf16_kernel(const float* __restrict__ in, bf16* __restrict__ out, int n) {
  int i = blockIdx.x * 256 + threadIdx.x;
  if (i < n) out[i] = (bf16)in[i];
}

// ---------------------------------------------------------------------------
// Causal depthwise conv (W=4) + SiLU; input (B,D,L) f32, output (B,L,D) bf16.
// Block: 256 threads, tile 64 d x 64 l, staged through LDS for coalescing.
// ---------------------------------------------------------------------------
__global__ __launch_bounds__(256)
void conv_silu_kernel(const float* __restrict__ x, const float* __restrict__ mask,
                      const float* __restrict__ info,
                      const float* __restrict__ wx, const float* __restrict__ bx,
                      const float* __restrict__ wm, const float* __restrict__ bm,
                      const float* __restrict__ wi, const float* __restrict__ bi,
                      bf16* __restrict__ u_bf, bf16* __restrict__ m_bf,
                      bf16* __restrict__ i_bf) {
  __shared__ float tile[64][68];
  int z = blockIdx.z;
  int b = z / 3, which = z % 3;
  const float* src; const float* wgt; const float* bia; bf16* dst;
  if (which == 0)      { src = x;    wgt = wx; bia = bx; dst = u_bf; }
  else if (which == 1) { src = mask; wgt = wm; bia = bm; dst = m_bf; }
  else                 { src = info; wgt = wi; bia = bi; dst = i_bf; }

  int d0 = blockIdx.y * 64, l0 = blockIdx.x * 64;
  int tid = threadIdx.x;
  for (int i = tid; i < 64 * 67; i += 256) {
    int di = i / 67, lj = i % 67;
    int gl = l0 - (WW - 1) + lj;
    tile[di][lj] = (gl >= 0) ? src[((size_t)b * DD + d0 + di) * LL + gl] : 0.f;
  }
  __syncthreads();

  int dOff = tid & 63;
  int d = d0 + dOff;
  float w0 = wgt[d * WW + 0], w1 = wgt[d * WW + 1];
  float w2 = wgt[d * WW + 2], w3 = wgt[d * WW + 3];
  float bc = bia[d];
  for (int li = (tid >> 6); li < 64; li += 4) {
    float a = bc + w0 * tile[dOff][li] + w1 * tile[dOff][li + 1]
                 + w2 * tile[dOff][li + 2] + w3 * tile[dOff][li + 3];
    float s = a / (1.f + __expf(-a));                       // SiLU
    dst[((size_t)b * LL + l0 + li) * DD + d] = (bf16)s;
  }
}

// ---------------------------------------------------------------------------
// WMMA bf16 NT GEMM:  Out[m,n] = sum_k A[m,k] * Bx[n,k]
// A: (M,K) bf16 row-major; Bx: (N,K) bf16 row-major (both K-contiguous).
// Each wave: MT M-tiles x 4 N-tiles.  K unrolled by 64 with two fragment
// sets (ping-pong) so loads and WMMAs overlap with NO register copies.
// Epilogues: 0 = store bf16 col-major (n*M+m)            (d_low)
//            1 = softplus(acc + bias[m]) -> f32 (n*M+m)  (dt)
//            2 = f32 (n*M+m)                             (Bm / Cm)
//            3 = f32 (B,D,L) + bias[m]                   (adjust -> d_out)
// ---------------------------------------------------------------------------
__device__ inline v16bf load_afrag(const bf16* __restrict__ aRow, int k0, int kh) {
  // lanes 0-15 (kh=0): K = {k0..k0+15}
  // lanes 16-31(kh=1): K = {k0+16..k0+31}
  v8bf lo = *(const v8bf*)(aRow + k0 + kh * 16);
  v8bf hi = *(const v8bf*)(aRow + k0 + kh * 16 + 8);
  return __builtin_shufflevector(lo, hi, 0, 1, 2, 3, 4, 5, 6, 7,
                                 8, 9, 10, 11, 12, 13, 14, 15);
}

template <int MT>
__device__ inline void load_frags(const bf16* __restrict__ aRow,
                                  const bf16* __restrict__ bRow, size_t K,
                                  int k0, int kh, v16bf (&a)[MT], v16bf (&b)[4]) {
  #pragma unroll
  for (int mt = 0; mt < MT; ++mt)
    a[mt] = load_afrag(aRow + (size_t)mt * 16 * K, k0, kh);
  #pragma unroll
  for (int t = 0; t < 4; ++t)
    b[t] = *(const v16bf*)(bRow + (size_t)t * 16 * K + k0);
}

template <int MT>
__device__ inline void mma_group(const v16bf (&a)[MT], const v16bf (&b)[4],
                                 v8f (&acc)[MT][4]) {
  #pragma unroll
  for (int t = 0; t < 4; ++t)
    #pragma unroll
    for (int mt = 0; mt < MT; ++mt)
      acc[mt][t] = __builtin_amdgcn_wmma_f32_16x16x32_bf16(
          false, a[mt], false, b[t], (short)0, acc[mt][t], false, false);
}

template <int EPI, int MT>
__global__ __launch_bounds__(256)
void gemm_nt_kernel(const bf16* __restrict__ A, const bf16* __restrict__ Bx,
                    int M, int N, int K,
                    float* __restrict__ outF, bf16* __restrict__ outB,
                    const float* __restrict__ bias) {
  int lane = threadIdx.x & 31;
  int wave = threadIdx.x >> 5;
  int m0 = blockIdx.y * (16 * MT);
  int n0 = blockIdx.x * 512 + wave * 64;
  int row = lane & 15, kh = lane >> 4;

  const bf16* aRow = A + (size_t)(m0 + row) * K;
  const bf16* bRow = Bx + (size_t)(n0 + row) * K + kh * 16;  // per-lane B stream
  v8f acc[MT][4];
  #pragma unroll
  for (int mt = 0; mt < MT; ++mt)
    #pragma unroll
    for (int t = 0; t < 4; ++t) acc[mt][t] = v8f{};

  // Two fragment sets, ping-pong (no copies).  K is a multiple of 64.
  v16bf aA[MT], bA[4], aB[MT], bB[4];
  load_frags<MT>(aRow, bRow, K, 0, kh, aA, bA);

  int k0 = 0;
  for (; k0 + 64 < K; k0 += 64) {
    load_frags<MT>(aRow, bRow, K, k0 + 32, kh, aB, bB);
    #pragma unroll
    for (int t = 0; t < 4; ++t)  // speculative stream prefetch, ~8 steps ahead
      __builtin_prefetch(bRow + (size_t)t * 16 * K + k0 + 288, 0, 3);
    mma_group<MT>(aA, bA, acc);
    load_frags<MT>(aRow, bRow, K, k0 + 64, kh, aA, bA);
    mma_group<MT>(aB, bB, acc);
  }
  // tail: one 64-deep double-step remains
  load_frags<MT>(aRow, bRow, K, k0 + 32, kh, aB, bB);
  mma_group<MT>(aA, bA, acc);
  mma_group<MT>(aB, bB, acc);

  int n = lane & 15;
  #pragma unroll
  for (int mt = 0; mt < MT; ++mt) {
    #pragma unroll
    for (int t = 0; t < 4; ++t) {
      int ncol = n0 + t * 16 + n;
      #pragma unroll
      for (int r = 0; r < 8; ++r) {
        int m = m0 + mt * 16 + kh * 8 + r;
        float v = acc[mt][t][r];
        if constexpr (EPI == 0) {
          outB[(size_t)ncol * M + m] = (bf16)v;
        } else if constexpr (EPI == 1) {
          v += bias[m];
          v = (v > 20.f) ? v : log1pf(__expf(v));             // softplus
          outF[(size_t)ncol * M + m] = v;
        } else if constexpr (EPI == 2) {
          outF[(size_t)ncol * M + m] = v;
        } else {
          int b = ncol / LL, l = ncol % LL;                   // M == DD here
          outF[((size_t)b * M + m) * LL + l] = v + bias[m];
        }
      }
    }
  }
}

// ---------------------------------------------------------------------------
// Chunked selective scan.
//  h_t = exp(dt_t*A) h_{t-1} + (dt_t*u_t) B_t ; y_t = <h_t, C_t>
// Pass1: per (b,d,chunk): P = prod(exp(dt*A)), S = local end-state (h0=0).
// Pass2: per (b,d,n): inter-chunk recurrence over the 64 chunks.
// Pass3: replay each chunk with the correct initial h; emit y (bf16, (B,L,D)).
// ---------------------------------------------------------------------------
__global__ __launch_bounds__(256)
void scan_pass1_kernel(const float* __restrict__ dt_f, const bf16* __restrict__ u_bf,
                       const float* __restrict__ Bv_f, const float* __restrict__ A_log,
                       float* __restrict__ chunkP, float* __restrict__ chunkS) {
  __shared__ float sB[LC][NN];
  int b = blockIdx.z, c = blockIdx.y, d0 = blockIdx.x * 256;
  int tid = threadIdx.x;
  int l0 = c * LC;
  for (int i = tid; i < LC * NN; i += 256)
    ((float*)sB)[i] = Bv_f[((size_t)b * LL + l0) * NN + i];
  __syncthreads();

  int d = d0 + tid;
  float Arow[NN];
  #pragma unroll
  for (int nn = 0; nn < NN; ++nn) Arow[nn] = -__expf(A_log[d * NN + nn]);

  float s[NN], p[NN];
  #pragma unroll
  for (int nn = 0; nn < NN; ++nn) { s[nn] = 0.f; p[nn] = 1.f; }

  for (int i = 0; i < LC; ++i) {
    size_t idx = ((size_t)b * LL + l0 + i) * DD + d;
    float dtv = dt_f[idx];
    float du = dtv * (float)u_bf[idx];
    #pragma unroll
    for (int nn = 0; nn < NN; ++nn) {
      float a = __expf(dtv * Arow[nn]);
      s[nn] = fmaf(a, s[nn], du * sB[i][nn]);
      p[nn] *= a;
    }
  }
  size_t o = (((size_t)b * NCH + c) * DD + d) * NN;
  #pragma unroll
  for (int nn = 0; nn < NN; ++nn) { chunkP[o + nn] = p[nn]; chunkS[o + nn] = s[nn]; }
}

__global__ __launch_bounds__(256)
void scan_pass2_kernel(const float* __restrict__ chunkP, const float* __restrict__ chunkS,
                       float* __restrict__ hstart) {
  size_t g = (size_t)blockIdx.x * 256 + threadIdx.x;   // over B*D*N
  size_t b = g / (DD * NN);
  size_t dn = g % (DD * NN);
  float h = 0.f;
  for (int c = 0; c < NCH; ++c) {
    size_t idx = ((b * NCH + c) * (size_t)(DD * NN)) + dn;
    hstart[idx] = h;
    h = fmaf(chunkP[idx], h, chunkS[idx]);
  }
}

__global__ __launch_bounds__(256)
void scan_pass3_kernel(const float* __restrict__ dt_f, const bf16* __restrict__ u_bf,
                       const float* __restrict__ Bv_f, const float* __restrict__ Cv_f,
                       const float* __restrict__ A_log, const float* __restrict__ hstart,
                       bf16* __restrict__ y_bf) {
  __shared__ float sB[LC][NN];
  __shared__ float sC[LC][NN];
  int b = blockIdx.z, c = blockIdx.y, d0 = blockIdx.x * 256;
  int tid = threadIdx.x;
  int l0 = c * LC;
  for (int i = tid; i < LC * NN; i += 256) {
    ((float*)sB)[i] = Bv_f[((size_t)b * LL + l0) * NN + i];
    ((float*)sC)[i] = Cv_f[((size_t)b * LL + l0) * NN + i];
  }
  __syncthreads();

  int d = d0 + tid;
  float Arow[NN];
  #pragma unroll
  for (int nn = 0; nn < NN; ++nn) Arow[nn] = -__expf(A_log[d * NN + nn]);

  float s[NN];
  size_t o = (((size_t)b * NCH + c) * DD + d) * NN;
  #pragma unroll
  for (int nn = 0; nn < NN; ++nn) s[nn] = hstart[o + nn];

  for (int i = 0; i < LC; ++i) {
    size_t idx = ((size_t)b * LL + l0 + i) * DD + d;
    float dtv = dt_f[idx];
    float du = dtv * (float)u_bf[idx];
    float y = 0.f;
    #pragma unroll
    for (int nn = 0; nn < NN; ++nn) {
      float a = __expf(dtv * Arow[nn]);
      s[nn] = fmaf(a, s[nn], du * sB[i][nn]);
      y = fmaf(s[nn], sC[i][nn], y);
    }
    y_bf[idx] = (bf16)y;
  }
}

// ---------------------------------------------------------------------------
// Host-side launcher
// ---------------------------------------------------------------------------
extern "C" void kernel_launch(void* const* d_in, const int* in_sizes, int n_in,
                              void* d_out, int out_size, void* d_ws, size_t ws_size,
                              hipStream_t stream) {
  const float* x       = (const float*)d_in[0];
  const float* mask    = (const float*)d_in[1];
  const float* info    = (const float*)d_in[2];
  const float* wx      = (const float*)d_in[3];
  const float* bx      = (const float*)d_in[4];
  const float* wm      = (const float*)d_in[5];
  const float* bm      = (const float*)d_in[6];
  const float* wi      = (const float*)d_in[7];
  const float* bi      = (const float*)d_in[8];
  const float* delta_w = (const float*)d_in[9];
  const float* dt_w    = (const float*)d_in[10];
  const float* dt_b    = (const float*)d_in[11];
  const float* B_w     = (const float*)d_in[12];
  const float* C_w     = (const float*)d_in[13];
  const float* A_log   = (const float*)d_in[14];
  const float* adj_w   = (const float*)d_in[15];
  const float* adj_b   = (const float*)d_in[16];
  float* out = (float*)d_out;

  char* base = (char*)d_ws;
  size_t off = 0;
  auto alloc = [&](size_t bytes) -> void* {
    void* p = base + off;
    off = (off + bytes + 255) & ~(size_t)255;
    return p;
  };
  const size_t BLD = (size_t)BB * LL * DD;
  bf16*  u_bf    = (bf16*)alloc(BLD * 2);
  bf16*  m_bf    = (bf16*)alloc(BLD * 2);
  bf16*  i_bf    = (bf16*)alloc(BLD * 2);
  bf16*  dlow_bf = (bf16*)alloc((size_t)BB * LL * RR * 2);
  float* dt_f    = (float*)alloc(BLD * 4);
  float* Bv_f    = (float*)alloc((size_t)BB * LL * NN * 4);
  float* Cv_f    = (float*)alloc((size_t)BB * LL * NN * 4);
  float* chunkP  = (float*)alloc((size_t)BB * NCH * DD * NN * 4);
  float* chunkS  = (float*)alloc((size_t)BB * NCH * DD * NN * 4);
  float* hstart  = (float*)alloc((size_t)BB * NCH * DD * NN * 4);
  bf16*  y_bf    = (bf16*)alloc(BLD * 2);
  bf16*  delta_w_bf = (bf16*)alloc((size_t)RR * DD * 2);
  bf16*  dt_w_bf    = (bf16*)alloc((size_t)DD * RR * 2);
  bf16*  B_w_bf     = (bf16*)alloc((size_t)NN * DD * 2);
  bf16*  C_w_bf     = (bf16*)alloc((size_t)NN * DD * 2);
  bf16*  adj_w_bf   = (bf16*)alloc((size_t)DD * DD * 2);

  // Weight conversions (tiny)
  f32_to_bf16_kernel<<<(RR * DD + 255) / 256, 256, 0, stream>>>(delta_w, delta_w_bf, RR * DD);
  f32_to_bf16_kernel<<<(DD * RR + 255) / 256, 256, 0, stream>>>(dt_w, dt_w_bf, DD * RR);
  f32_to_bf16_kernel<<<(NN * DD + 255) / 256, 256, 0, stream>>>(B_w, B_w_bf, NN * DD);
  f32_to_bf16_kernel<<<(NN * DD + 255) / 256, 256, 0, stream>>>(C_w, C_w_bf, NN * DD);
  f32_to_bf16_kernel<<<(DD * DD + 255) / 256, 256, 0, stream>>>(adj_w, adj_w_bf, DD * DD);

  // Causal dwconv + SiLU (x, mask, info), transposed outputs (B,L,D) bf16
  conv_silu_kernel<<<dim3(LL / 64, DD / 64, BB * 3), 256, 0, stream>>>(
      x, mask, info, wx, bx, wm, bm, wi, bi, u_bf, m_bf, i_bf);

  const int BL = BB * LL;  // 16384 columns

  // d_low = delta_w @ m        : M=64,  K=1024  (2 m-tiles/wave)
  gemm_nt_kernel<0, 2><<<dim3(BL / 512, RR / 32), 256, 0, stream>>>(
      delta_w_bf, m_bf, RR, BL, DD, nullptr, dlow_bf, nullptr);
  // dt = softplus(dt_w @ d_low + dt_b) : M=1024, K=64
  gemm_nt_kernel<1, 2><<<dim3(BL / 512, DD / 32), 256, 0, stream>>>(
      dt_w_bf, dlow_bf, DD, BL, RR, dt_f, nullptr, dt_b);
  // Bm = B_w @ inf             : M=16, K=1024
  gemm_nt_kernel<2, 1><<<dim3(BL / 512, 1), 256, 0, stream>>>(
      B_w_bf, i_bf, NN, BL, DD, Bv_f, nullptr, nullptr);
  // Cm = C_w @ u               : M=16, K=1024
  gemm_nt_kernel<2, 1><<<dim3(BL / 512, 1), 256, 0, stream>>>(
      C_w_bf, u_bf, NN, BL, DD, Cv_f, nullptr, nullptr);

  // Chunked selective scan
  scan_pass1_kernel<<<dim3(DD / 256, NCH, BB), 256, 0, stream>>>(
      dt_f, u_bf, Bv_f, A_log, chunkP, chunkS);
  scan_pass2_kernel<<<dim3((BB * DD * NN) / 256), 256, 0, stream>>>(
      chunkP, chunkS, hstart);
  scan_pass3_kernel<<<dim3(DD / 256, NCH, BB), 256, 0, stream>>>(
      dt_f, u_bf, Bv_f, Cv_f, A_log, hstart, y_bf);

  // out = adj_w @ y + adj_b    : M=1024, K=1024 (dominant GEMM, 2 m-tiles/wave)
  gemm_nt_kernel<3, 2><<<dim3(BL / 512, DD / 32), 256, 0, stream>>>(
      adj_w_bf, y_bf, DD, BL, DD, out, nullptr, adj_b);
}
